// ManifoldHead_81475529605937
// MI455X (gfx1250) — matrix operand order repaired and checked by reading, not compiled, not run on hardware
//
#include <hip/hip_runtime.h>
#include <hip/hip_bf16.h>
#include <math.h>

// ---------------------------------------------------------------------------
// ManifoldHead forward, fused per-cloud, CDNA5 (gfx1250) wave32 + WMMA f16.
// L=6,B=16,V=50 -> N=4800 clouds, P=20 points, K=4 knn, feat dim 800.
// Weights pre-swizzled into WMMA fragment-major layout (32B/lane contiguous).
// ---------------------------------------------------------------------------

#define PP   20
#define KNN_  4

typedef __attribute__((ext_vector_type(16))) _Float16 v16h;
typedef __attribute__((ext_vector_type(8)))  _Float16 v8h;
typedef __attribute__((ext_vector_type(8)))  float    v8f;

// ---------------- workspace layout (bytes) ----------------
// edgeW : _Float16 [36][4 frag][32 lane][16]  (fragment-major) @ 0      (147456 B)
// edgeB : float    [36][32]                                   @ 147456 (4608 B)
// W2h   : _Float16 [128 frag][32 lane][16]    (fragment-major) @ 152064 (131072 B)
#define WS_EDGEW 0
#define WS_EDGEB 147456
#define WS_W2H   152064

struct PrepArgs {
  const float* W[36];   // edge layer weights, layer id = (u*4+conv)*3+l
  const float* B[36];   // edge layer biases
  const float* W2;      // gcn layer-2 weight [256][256]
};

struct MainArgs {
  const float* __restrict__ pts;       // [N][20][2]
  const float* __restrict__ clsPrior;  // [N][3]
  const float* __restrict__ emb;       // [N][20][256]
  const float* __restrict__ gcnW1;     // [2][256]
  const float* __restrict__ gcnB1;     // [256]
  const float* __restrict__ gcnB2;     // [256]
  const float* __restrict__ poolP;     // [800]
  const float* __restrict__ adjW;      // [800][2]
  const float* __restrict__ adjB;      // [2]
  const float* __restrict__ clsW1;     // [803][256]
  const float* __restrict__ clsB1;     // [256]
  const float* __restrict__ clsW2;     // [256][3]
  const float* __restrict__ clsB2;     // [3]
  const _Float16* __restrict__ edgeW;    // packed fragment-major
  const float*    __restrict__ edgeBias; // packed [36][32]
  const _Float16* __restrict__ W2h;      // packed fragment-major
  float* __restrict__ outCls;
  float* __restrict__ outBbox;
  float* __restrict__ outPts;
  int nCloud;
};

// ---------------- weight pre-pack: f32 -> f16, fragment-major --------------
// B fragment layout (32x16 f16, per CDNA5 ISA): lane -> (hf=lane>>4, n=lane&15),
// element e (0..15) -> K = ksBase + 8*hf + (e&7) + 16*(e>>3), column n. Stored contiguously
// per lane so a fragment load is one 32-byte v16h access.
__global__ void prep_kernel(PrepArgs p, _Float16* edgeW, float* edgeB, _Float16* W2h) {
  const int b = blockIdx.x, t = threadIdx.x;
  if (b < 36) {
    const int l = b % 3, conv = (b / 3) % 4;
    const int Kin  = (l == 0) ? ((conv == 0) ? 4 : 48) : 32;
    const int Nout = (l == 2) ? 24 : 32;
    const float* W = p.W[b];
    _Float16* dst = edgeW + b * 2048;   // 4 frags x 32 lanes x 16
    for (int i = t; i < 2048; i += blockDim.x) {
      const int e = i & 15, lane = (i >> 4) & 31, frag = i >> 9; // frag=ks*2+nt
      const int ks = frag >> 1, nt = frag & 1;
      const int hf = lane >> 4, n = lane & 15;
      const int k = ks * 32 + 8 * hf + (e & 7) + ((e >> 3) << 4);
      const int col = nt * 16 + n;
      const float v = (k < Kin && col < Nout) ? W[k * Nout + col] : 0.0f;
      dst[i] = (_Float16)v;
    }
    if (t < 32) edgeB[b * 32 + t] = (t < Nout) ? p.B[b][t] : 0.0f;
  } else {
    const int base = (b - 36) * 1024;   // 64 blocks cover 65536 elems
    for (int j = t; j < 1024; j += blockDim.x) {
      const int i = base + j;
      const int e = i & 15, lane = (i >> 4) & 31, frag = i >> 9; // frag=nt*8+ks
      const int ks = frag & 7, nt = frag >> 3;
      const int hf = lane >> 4, n = lane & 15;
      const int k = ks * 32 + 8 * hf + (e & 7) + ((e >> 3) << 4);
      const int col = nt * 16 + n;
      W2h[i] = (_Float16)p.W2[k * 256 + col];
    }
  }
}

// ---------------- WMMA fragment helpers ------------------------------------
// A (16x32 f16) from row-major src: per lane, elements are two contiguous
// 16-byte chunks: [m*ld + 8*hf .. +7] and [m*ld + 16 + 8*hf .. +7].
__device__ __forceinline__ v16h fragA(const _Float16* s, int ld, int lane) {
  const int hf = lane >> 4, m = lane & 15;
  const v8h lo = *(const v8h*)(s + m * ld + 8 * hf);
  const v8h hi = *(const v8h*)(s + m * ld + 16 + 8 * hf);
  v16h a;
#pragma unroll
  for (int i = 0; i < 8; ++i) { a[i] = lo[i]; a[8 + i] = hi[i]; }
  return a;
}
// B fragment from pre-swizzled fragment-major storage: one 32B load per lane.
__device__ __forceinline__ v16h fragBP(const _Float16* s, int lane) {
  return *(const v16h*)(s + lane * 16);
}
__device__ __forceinline__ v8f wmma_f16(v16h a, v16h b, v8f c) {
  return __builtin_amdgcn_wmma_f32_16x16x32_f16(false, a, false, b, (short)0, c,
                                                false, false);
}

// One edge-MLP layer: D[80][32] = relu(A[80][lda] @ W + bias)
// Wp: fragment-major [frag = ks*2+nt][32][16]; B frags hoisted to registers.
__device__ void edge_mlp_layer(const _Float16* Abuf, int lda, int ksteps,
                               const _Float16* Wp, const float* bias,
                               _Float16* Dbuf, int lane) {
  const v16h b00 = fragBP(Wp + 0 * 512, lane);
  const v16h b01 = fragBP(Wp + 1 * 512, lane);
  v16h b10 = {}, b11 = {};
  if (ksteps == 2) { b10 = fragBP(Wp + 2 * 512, lane); b11 = fragBP(Wp + 3 * 512, lane); }
  const int nb = lane & 15;
  const float bb0 = bias[nb], bb1 = bias[16 + nb];
#pragma unroll
  for (int mt = 0; mt < 5; ++mt) {
    const v16h a0 = fragA(Abuf + (mt * 16) * lda, lda, lane);
    v8f c0 = {}, c1 = {};
    c0 = wmma_f16(a0, b00, c0);
    c1 = wmma_f16(a0, b01, c1);
    if (ksteps == 2) {
      const v16h a1 = fragA(Abuf + (mt * 16) * lda + 32, lda, lane);
      c0 = wmma_f16(a1, b10, c0);
      c1 = wmma_f16(a1, b11, c1);
    }
    const int mb = mt * 16 + 8 * (lane >> 4);
#pragma unroll
    for (int r = 0; r < 8; ++r) {
      const float v0 = c0[r] + bb0;
      const float v1 = c1[r] + bb1;
      Dbuf[(mb + r) * 32 + nb]      = (_Float16)(v0 > 0.0f ? v0 : 0.0f);
      Dbuf[(mb + r) * 32 + 16 + nb] = (_Float16)(v1 > 0.0f ? v1 : 0.0f);
    }
  }
  __syncthreads();
}

// knn (4 smallest squared distances, self excluded, ties -> lower index)
__device__ void knn20(const float* f /*[20][48]*/, int Cin, int lane,
                      float* dist /*[20*20]*/, float* nrm /*[20]*/,
                      int* knnI /*[20*4]*/) {
  if (lane < PP) {
    float s = 0.f;
    for (int c = 0; c < Cin; ++c) { float v = f[lane * 48 + c]; s += v * v; }
    nrm[lane] = s;
  }
  __syncthreads();
  if (lane < PP) {
    for (int q = 0; q < PP; ++q) {
      float dot = 0.f;
      for (int c = 0; c < Cin; ++c) dot += f[lane * 48 + c] * f[q * 48 + c];
      float d = nrm[lane] + nrm[q] - 2.f * dot;
      if (q == lane) d += 1e9f;
      dist[lane * PP + q] = d;
    }
    unsigned taken = 0;
    for (int k = 0; k < KNN_; ++k) {
      float best = 3.0e38f; int bi = 0;
      for (int q = 0; q < PP; ++q) {
        if (taken & (1u << q)) continue;
        const float d = dist[lane * PP + q];
        if (d < best) { best = d; bi = q; }
      }
      taken |= (1u << bi);
      knnI[lane * KNN_ + k] = bi;
    }
  }
  __syncthreads();
}

// ---------------- fused per-cloud kernel: 1 block = 1 wave = 1 cloud ------
__global__ void __launch_bounds__(32) manifold_main(MainArgs A) {
  const int cloud = blockIdx.x;
  if (cloud >= A.nCloud) return;
  const int lane = threadIdx.x;

  extern __shared__ char smem[];
  float* feat   = (float*)(smem + 0);       // [20][800]
  float* fcur   = (float*)(smem + 64000);   // [20][48]
  float* ptsS   = (float*)(smem + 67840);   // [20][2]
  float* tS     = (float*)(smem + 68000);   // [20][2]
  float* nrmS   = (float*)(smem + 68160);   // [20]
  float* scoreS = (float*)(smem + 68240);   // [20]
  float* valsS  = (float*)(smem + 68320);   // [20]
  float* redS   = (float*)(smem + 68400);   // [4]
  int*   knnS   = (int*)  (smem + 68416);   // [20][4]
  int*   pidxS  = (int*)  (smem + 68736);   // [20]
  float* outP   = (float*)(smem + 68816);   // [20][2]
  float* distS  = (float*)(smem + 68992);   // [20][20]
  float* AhatS  = (float*)(smem + 70592);   // [20][20]
  float* gmax   = (float*)(smem + 72192);   // [800]
  float* o1     = (float*)(smem + 75392);   // [256]
  float* hS     = (float*)(smem + 76416);   // [20][256]
  _Float16* eF   = (_Float16*)(smem + 96896);   // [80][64]
  _Float16* act0 = (_Float16*)(smem + 107136);  // [80][32]
  _Float16* act1 = (_Float16*)(smem + 112256);  // [80][32]
  _Float16* wL   = (_Float16*)(smem + 117376);  // [3][2048] fragment-major
  _Float16* g2h  = (_Float16*)(smem + 129664);  // [32][256]
  // total dynamic LDS = 146048 B

  for (int i = lane; i < PP * 2; i += 32)
    ptsS[i] = A.pts[(size_t)cloud * PP * 2 + i];
  __builtin_prefetch(A.emb + (size_t)cloud * PP * 256 + lane * 160, 0, 0);
  __syncthreads();

  // ===================== edge branch: 3 units x 4 EdgeConvs ================
  for (int u = 0; u < 3; ++u) {
    if (lane < PP) { fcur[lane * 48 + 0] = ptsS[lane * 2 + 0];
                     fcur[lane * 48 + 1] = ptsS[lane * 2 + 1]; }
    __syncthreads();
    int Cin = 2;
    for (int cv = 0; cv < 4; ++cv) {
      const int lid0 = (u * 4 + cv) * 3;
      { // stage this conv's 3 packed weight layers into LDS (128-bit copies)
        const ulong2* src = (const ulong2*)(A.edgeW + lid0 * 2048);
        ulong2* dst = (ulong2*)wL;
        for (int i = lane; i < 768; i += 32) dst[i] = src[i];
      }
      knn20(fcur, Cin, lane, distS, nrmS, knnS);
      { // zero edge-feature buffer (padding cols must stay 0)
        ulong2* z = (ulong2*)eF;
        const ulong2 zz = {0ul, 0ul};
        for (int i = lane; i < (80 * 64) / 8; i += 32) z[i] = zz;
      }
      __syncthreads();
      // e[p*4+kk] = [ f[j]-f[p] , f[p] ]
      for (int i = lane; i < 80 * Cin; i += 32) {
        const int r = i / Cin, c = i - r * Cin;
        const int p = r >> 2, kk = r & 3;
        const int j = knnS[p * KNN_ + kk];
        const float fi = fcur[p * 48 + c], fj = fcur[j * 48 + c];
        eF[r * 64 + c]       = (_Float16)(fj - fi);
        eF[r * 64 + Cin + c] = (_Float16)fi;
      }
      __syncthreads();
      const int ks0 = (2 * Cin + 31) / 32;
      edge_mlp_layer(eF,  64, ks0, wL + 0 * 2048, A.edgeBias + (lid0 + 0) * 32, act0, lane);
      edge_mlp_layer(act0, 32, 1,  wL + 1 * 2048, A.edgeBias + (lid0 + 1) * 32, act1, lane);
      edge_mlp_layer(act1, 32, 1,  wL + 2 * 2048, A.edgeBias + (lid0 + 2) * 32, act0, lane);
      // max over k neighbors -> next feature / output slot
      if (lane < PP) {
        for (int j = 0; j < 24; ++j) {
          float m =           (float)act0[(lane * 4 + 0) * 32 + j];
          m = fmaxf(m, (float)act0[(lane * 4 + 1) * 32 + j]);
          m = fmaxf(m, (float)act0[(lane * 4 + 2) * 32 + j]);
          m = fmaxf(m, (float)act0[(lane * 4 + 3) * 32 + j]);
          fcur[lane * 48 + j] = m;
          feat[lane * 800 + u * 96 + cv * 24 + j] = m;
        }
      }
      __syncthreads();
      Cin = 24;
    }
  }

  // ===================== GCN branch ========================================
  if (lane < PP) { fcur[lane * 48 + 0] = ptsS[lane * 2 + 0];
                   fcur[lane * 48 + 1] = ptsS[lane * 2 + 1]; }
  __syncthreads();
  knn20(fcur, 2, lane, distS, nrmS, knnS);
  if (lane < PP) {          // A = max(A, A^T) + I ; dinv = rsqrt(rowsum)
    float rs = 0.f;
    for (int q = 0; q < PP; ++q) {
      float a1 = 0.f, a2 = 0.f;
      for (int k = 0; k < KNN_; ++k) {
        if (knnS[lane * KNN_ + k] == q) a1 = 1.f;
        if (knnS[q * KNN_ + k] == lane) a2 = 1.f;
      }
      const float a = fmaxf(a1, a2) + ((q == lane) ? 1.f : 0.f);
      distS[lane * PP + q] = a;
      rs += a;
    }
    nrmS[lane] = 1.f / sqrtf(rs);
  }
  __syncthreads();
  if (lane < PP) {          // Ahat, and t = Ahat @ pts
    float tx = 0.f, ty = 0.f;
    for (int q = 0; q < PP; ++q) {
      const float ah = distS[lane * PP + q] * nrmS[lane] * nrmS[q];
      AhatS[lane * PP + q] = ah;
      tx += ah * ptsS[q * 2 + 0];
      ty += ah * ptsS[q * 2 + 1];
    }
    tS[lane * 2 + 0] = tx; tS[lane * 2 + 1] = ty;
  }
  __syncthreads();
  for (int i = lane; i < PP * 256; i += 32) {   // h = relu(t @ W1 + b1)
    const int p = i >> 8, c = i & 255;
    const float v = tS[p * 2] * A.gcnW1[c] + tS[p * 2 + 1] * A.gcnW1[256 + c] + A.gcnB1[c];
    hS[i] = (v > 0.f) ? v : 0.f;
  }
  { ulong2* z = (ulong2*)g2h;
    const ulong2 zz = {0ul, 0ul};
    for (int i = lane; i < (32 * 256) / 8; i += 32) z[i] = zz; }
  __syncthreads();
  for (int i = lane; i < PP * 256; i += 32) {   // g2 = Ahat @ h  (f16, padded)
    const int p = i >> 8, c = i & 255;
    float s = 0.f;
    for (int q = 0; q < PP; ++q) s += AhatS[p * PP + q] * hS[q * 256 + c];
    g2h[p * 256 + c] = (_Float16)s;
  }
  __syncthreads();
  // feat_gcn = g2 @ W2 + b2 : WMMA, M=32(2 tiles), N=256(16), K=256(8 steps).
  // B fragments (L2-resident, fragment-major) hoisted and reused by both M tiles.
  for (int nt = 0; nt < 16; ++nt) {
    v16h bf[8];
#pragma unroll
    for (int ks = 0; ks < 8; ++ks)
      bf[ks] = fragBP(A.W2h + (nt * 8 + ks) * 512, lane);
    const int n  = nt * 16 + (lane & 15);
    const float bb = A.gcnB2[n];
#pragma unroll
    for (int mt = 0; mt < 2; ++mt) {
      v8f c = {};
#pragma unroll
      for (int ks = 0; ks < 8; ++ks) {
        const v16h a = fragA(g2h + (mt * 16) * 256 + ks * 32, 256, lane);
        c = wmma_f16(a, bf[ks], c);
      }
      const int mb = mt * 16 + 8 * (lane >> 4);
#pragma unroll
      for (int r = 0; r < 8; ++r) {
        const int m = mb + r;
        if (m < PP) feat[m * 800 + 288 + n] = c[r] + bb;
      }
    }
  }
  __syncthreads();

  // ===================== decoder embedding =================================
  for (int i = lane; i < PP * 256; i += 32) {
    const int p = i >> 8, c = i & 255;
    feat[p * 800 + 544 + c] = A.emb[(size_t)cloud * PP * 256 + i];
  }
  __syncthreads();

  // ===================== ClassReg ==========================================
  for (int j = lane; j < 800; j += 32) {
    float m = feat[j];
    for (int p = 1; p < PP; ++p) m = fmaxf(m, feat[p * 800 + j]);
    gmax[j] = m;
  }
  __syncthreads();
  for (int c = lane; c < 256; c += 32) {
    float s = A.clsB1[c];
    for (int j = 0; j < 800; ++j) s += gmax[j] * A.clsW1[j * 256 + c];
    for (int i = 0; i < 3; ++i)
      s += A.clsPrior[cloud * 3 + i] * A.clsW1[(800 + i) * 256 + c];
    o1[c] = (s > 0.f) ? s : 0.f;
  }
  __syncthreads();
  if (lane < 3) {
    float s = A.clsB2[lane];
    for (int c = 0; c < 256; ++c) s += o1[c] * A.clsW2[c * 3 + lane];
    A.outCls[cloud * 3 + lane] = s;
  }

  // ===================== gPool reorder + position adjust ===================
  if (lane == 0) {
    float s = 0.f;
    for (int j = 0; j < 800; ++j) s += A.poolP[j] * A.poolP[j];
    redS[0] = sqrtf(s);
  }
  __syncthreads();
  if (lane < PP) {
    float s = 0.f;
    for (int j = 0; j < 800; ++j) s += feat[lane * 800 + j] * A.poolP[j];
    scoreS[lane] = s / redS[0];
  }
  __syncthreads();
  if (lane < PP) {   // rank-sort descending, ties -> lower original index
    const float sv = scoreS[lane];
    int rank = 0;
    for (int q = 0; q < PP; ++q) {
      const float sq = scoreS[q];
      if (sq > sv || (sq == sv && q < lane)) rank++;
    }
    pidxS[rank] = lane;
    valsS[rank] = sv;
  }
  __syncthreads();
  if (lane < PP) {
    const int src = pidxS[lane];
    const float gate = 1.f / (1.f + expf(-valsS[lane]));
    float ax = A.adjB[0], ay = A.adjB[1];
    for (int j = 0; j < 800; ++j) {
      const float fv = feat[src * 800 + j] * gate;
      ax += fv * A.adjW[j * 2 + 0];
      ay += fv * A.adjW[j * 2 + 1];
    }
    const float ox = ptsS[src * 2 + 0] + tanhf(ax);
    const float oy = ptsS[src * 2 + 1] + tanhf(ay);
    outP[lane * 2 + 0] = ox; outP[lane * 2 + 1] = oy;
    A.outPts[(size_t)cloud * PP * 2 + lane * 2 + 0] = ox;
    A.outPts[(size_t)cloud * PP * 2 + lane * 2 + 1] = oy;
  }
  __syncthreads();
  if (lane == 0) {   // transform_box 'minmax'
    float xmn = outP[0], xmx = outP[0], ymn = outP[1], ymx = outP[1];
    for (int r = 1; r < PP; ++r) {
      xmn = fminf(xmn, outP[r * 2]);     xmx = fmaxf(xmx, outP[r * 2]);
      ymn = fminf(ymn, outP[r * 2 + 1]); ymx = fmaxf(ymx, outP[r * 2 + 1]);
    }
    float* bb = A.outBbox + cloud * 4;
    bb[0] = (xmn + xmx) * 0.5f; bb[1] = (ymn + ymx) * 0.5f;
    bb[2] = xmx - xmn;          bb[3] = ymx - ymn;
  }
}

// ---------------------------------------------------------------------------
extern "C" void kernel_launch(void* const* d_in, const int* in_sizes, int n_in,
                              void* d_out, int out_size, void* d_ws, size_t ws_size,
                              hipStream_t stream) {
  int idx = 0;
  const float* pts  = (const float*)d_in[idx++];   // all_pts_preds
  const float* clsP = (const float*)d_in[idx++];   // all_cls_scores
  const float* emb  = (const float*)d_in[idx++];   // decoder_embed

  PrepArgs pa;
  for (int i = 0; i < 36; ++i) {                    // params['edge'] (W,b)*36
    pa.W[i] = (const float*)d_in[idx++];
    pa.B[i] = (const float*)d_in[idx++];
  }
  const float* gW1 = (const float*)d_in[idx++];     // params['gcn']
  const float* gB1 = (const float*)d_in[idx++];
  const float* gW2 = (const float*)d_in[idx++];
  const float* gB2 = (const float*)d_in[idx++];
  const float* poolP = (const float*)d_in[idx++];   // params['pool_p']
  const float* adjW  = (const float*)d_in[idx++];   // params['pool_adj']
  const float* adjB  = (const float*)d_in[idx++];
  const float* cW1 = (const float*)d_in[idx++];     // params['cls']
  const float* cB1 = (const float*)d_in[idx++];
  const float* cW2 = (const float*)d_in[idx++];
  const float* cB2 = (const float*)d_in[idx++];
  pa.W2 = gW2;

  _Float16* edgeW = (_Float16*)((char*)d_ws + WS_EDGEW);
  float*    edgeB = (float*)   ((char*)d_ws + WS_EDGEB);
  _Float16* W2h   = (_Float16*)((char*)d_ws + WS_W2H);

  prep_kernel<<<100, 256, 0, stream>>>(pa, edgeW, edgeB, W2h);

  const int nCloud = in_sizes[0] / (PP * 2);        // 4800
  float* out = (float*)d_out;

  MainArgs ma;
  ma.pts = pts; ma.clsPrior = clsP; ma.emb = emb;
  ma.gcnW1 = gW1; ma.gcnB1 = gB1; ma.gcnB2 = gB2;
  ma.poolP = poolP; ma.adjW = adjW; ma.adjB = adjB;
  ma.clsW1 = cW1; ma.clsB1 = cB1; ma.clsW2 = cW2; ma.clsB2 = cB2;
  ma.edgeW = edgeW; ma.edgeBias = edgeB; ma.W2h = W2h;
  ma.outCls  = out;
  ma.outBbox = out + (size_t)nCloud * 3;
  ma.outPts  = out + (size_t)nCloud * 7;
  ma.nCloud = nCloud;

  manifold_main<<<nCloud, 32, 146048, stream>>>(ma);
}
